// HarmonicSynthesizer_39857296507154
// MI455X (gfx1250) — matrix operand order repaired and checked by reading, not compiled, not run on hardware
//
#include <hip/hip_runtime.h>
#include <hip/hip_bf16.h>
#include <math.h>

typedef __attribute__((ext_vector_type(16))) _Float16 v16h;
typedef __attribute__((ext_vector_type(8)))  _Float16 v8h;
typedef __attribute__((ext_vector_type(8)))  float    v8f;

#define T_FRAMES 500
#define N_HARM   100
#define HOP      160
#define L_SAMP   (T_FRAMES * HOP)        /* 80000 samples per batch   */
#define N_RUNS   (L_SAMP / 16)           /* 5000 16-sample runs/tiles */
#define NT_PER_WAVE 10
#define BLOCKS_PER_BATCH (N_RUNS / (4 * NT_PER_WAVE))   /* 125 */

#define SCALE_POS ((float)(T_FRAMES - 1) / (float)(L_SAMP - 1))
#define INV_SR    (1.0f / 16000.0f)

// -------- piecewise-linear f0 upsample (align_corners=True) --------
__device__ __forceinline__ float f0_up(const float* __restrict__ f0b, int s) {
  float pos = (float)s * SCALE_POS;
  int lo = (int)pos;
  int hi = min(lo + 1, T_FRAMES - 1);
  float w = pos - (float)lo;
  float a = f0b[lo];
  float c = f0b[hi];
  return fmaf(w, c - a, a);
}

__device__ __forceinline__ double run_sum16(const float* __restrict__ f0b, int run) {
  double acc = 0.0;
  int s0 = run * 16;
  #pragma unroll 4
  for (int i = 0; i < 16; ++i) acc += (double)f0_up(f0b, s0 + i);
  return acc * (1.0 / 16000.0);
}

// ================== Kernel 1: hierarchical prefix of f0_up/SR ==================
// prefFrac[b][run] = frac( sum_{l < 16*run} f0_up[b][l]/SR )  (exclusive, double scan)
__global__ void __launch_bounds__(64)
harm_prefix_kernel(const float* __restrict__ f0, float* __restrict__ prefFrac) {
  __shared__ double sSum[50];
  __shared__ double sPref[50];
  const int b = blockIdx.x;
  const int t = threadIdx.x;
  const float* f0b = f0 + b * T_FRAMES;

  if (t < 50) {                       // 50 super-chunks of 100 runs each
    double acc = 0.0;
    for (int j = 0; j < 100; ++j) acc += run_sum16(f0b, t * 100 + j);
    sSum[t] = acc;
  }
  __syncthreads();
  if (t == 0) {                       // serial double scan of 50 values
    double a = 0.0;
    for (int i = 0; i < 50; ++i) { sPref[i] = a; a += sSum[i]; }
  }
  __syncthreads();
  if (t < 50) {                       // expand: exclusive prefix per run, store frac
    double acc = sPref[t];
    for (int j = 0; j < 100; ++j) {
      int run = t * 100 + j;
      prefFrac[b * N_RUNS + run] = (float)(acc - floor(acc));
      acc += run_sum16(f0b, run);
    }
  }
}

// ================== Kernel 2: oscillator bank + WMMA dot products ==================
__global__ void __launch_bounds__(128)
harm_synth_kernel(const float* __restrict__ f0,
                  const float* __restrict__ amps,
                  const float* __restrict__ prefFrac,
                  float* __restrict__ out) {
  // per-wave private LDS: [16 samples][128 harmonics] f16 for sin and for amps
  __shared__ __align__(32) _Float16 smem[4 * 4096];

  const int lane = threadIdx.x & 31;
  const int wave = threadIdx.x >> 5;
  const int b    = blockIdx.x / BLOCKS_PER_BATCH;
  const int seg  = blockIdx.x % BLOCKS_PER_BATCH;

  const float* __restrict__ f0b  = f0   + b * T_FRAMES;
  const float* __restrict__ ampb = amps + b * T_FRAMES * N_HARM;
  float* __restrict__ outb       = out  + b * L_SAMP;

  _Float16* sinT = smem + wave * 4096;        // 16*128
  _Float16* ampT = sinT + 2048;               // 16*128

  const int sIdx  = lane & 15;                // sample row / B column for this lane
  const int hsel  = lane >> 4;                // which harmonic half this lane generates
  const int kstart = hsel * 64;
  const int h0     = 1 + hsel * 64;           // first harmonic of this lane's run

  // diagonal-extraction constants (uniform over the whole kernel)
  const bool isDiag = (lane < 8) || (lane >= 24);
  const int  vi     = (lane < 8) ? lane : (lane - 24);   // accumulator VGPR index
  const int  di     = (lane < 8) ? lane : (lane - 16);   // diagonal / sample index

  for (int t = 0; t < NT_PER_WAVE; ++t) {
    const int tile = seg * (4 * NT_PER_WAVE) + wave * NT_PER_WAVE + t;  // 0..4999
    const int s0   = tile * 16;
    const int s    = s0 + sIdx;

    // ---- per-sample inclusive phase P = prefFrac(run) + shuffle-scan prefix ----
    // each lane evaluates exactly one f0_up; 4-step inclusive scan over 16-lane segs
    float sc = f0_up(f0b, s) * INV_SR;
    #pragma unroll
    for (int d = 1; d < 16; d <<= 1) {
      float o = __shfl_up(sc, (unsigned)d, 16);
      sc += (sIdx >= d) ? o : 0.0f;
    }
    const float p = prefFrac[b * N_RUNS + tile] + sc;

    // ---- amplitude interpolation params for this sample ----
    const float pos = (float)s * SCALE_POS;
    const int lo = (int)pos;
    const int hi = min(lo + 1, T_FRAMES - 1);
    const float w = pos - (float)lo;
    const float* __restrict__ aLo = ampb + lo * N_HARM;
    const float* __restrict__ aHi = ampb + hi * N_HARM;

    // ---- oscillator seed: sin(2*pi*h*P) = sin(2*pi*h*frac(P)) for integer h ----
    // v_sin_f32 / v_cos_f32 take revolutions (input pre-scaled by 1/2pi)
    float pf = p - floorf(p);
    float s1 = __builtin_amdgcn_sinf(pf);
    float c1 = __builtin_amdgcn_cosf(pf);
    float q  = (float)h0 * p;
    q -= floorf(q);
    float sh = __builtin_amdgcn_sinf(q);
    float ch = __builtin_amdgcn_cosf(q);

    // ---- generate 64 harmonics (sin + interpolated amp) into LDS tiles ----
    _Float16* srow = sinT + sIdx * 128 + kstart;
    _Float16* arow = ampT + sIdx * 128 + kstart;
    for (int g = 0; g < 8; ++g) {
      v8h sv, av;
      #pragma unroll
      for (int i2 = 0; i2 < 8; ++i2) {
        const int k = kstart + g * 8 + i2;     // channel = harmonic-1
        sv[i2] = (_Float16)sh;
        const int kc = min(k, N_HARM - 1);     // clamped, branchless pad to 128
        float loV = aLo[kc];
        float hiV = aHi[kc];
        float a_f = (k < N_HARM) ? fmaf(w, hiV - loV, loV) : 0.0f;
        av[i2] = (_Float16)a_f;
        // rotate phasor by one harmonic step
        float ns = fmaf(sh, c1, ch * s1);
        float nc = fmaf(ch, c1, -sh * s1);
        sh = ns; ch = nc;
      }
      *(v8h*)(srow + g * 8) = sv;
      *(v8h*)(arow + g * 8) = av;
    }

    // same-wave LDS RAW: make stores visible before fragment loads
    asm volatile("s_wait_dscnt 0x0" ::: "memory");

    // ---- 4x v_wmma_f32_16x16x32_f16 : D = Sin(16x128) x AmpU^T(128x16) ----
    v8f cAcc = {};
    const _Float16* srAll = sinT + sIdx * 128;  // A row M = lane%16
    const _Float16* brAll = ampT + sIdx * 128;  // B col N = lane%16
    #pragma unroll
    for (int kb = 0; kb < 128; kb += 32) {
      v16h aF = *(const v16h*)(srAll + kb + 16 * hsel);     // K = kb+16*hsel+[0..15]
      v16h bF = *(const v16h*)(brAll + kb + 16 * hsel);     // K = kb+16*hsel+[0..15]
      cAcc = __builtin_amdgcn_wmma_f32_16x16x32_f16(
          /*neg_a=*/false, aF, /*neg_b=*/false, bF,
          /*c_mod=*/(short)0, cAcc, /*reuse_a=*/false, /*reuse_b=*/false);
    }

    // ---- extract diagonal D[i][i] = audio[s0+i], single predicated store ----
    // lanes 0..7: diag i=lane in VGPR[lane]; lanes 24..31: diag i=lane-16 in VGPR[lane-24]
    float v = 0.0f;
    #pragma unroll
    for (int r = 0; r < 8; ++r) v = (vi == r) ? cAcc[r] : v;
    if (isDiag) outb[s0 + di] = v;
  }
}

// ============================ launch ============================
extern "C" void kernel_launch(void* const* d_in, const int* in_sizes, int n_in,
                              void* d_out, int out_size, void* d_ws, size_t ws_size,
                              hipStream_t stream) {
  const float* f0   = (const float*)d_in[0];   // [B, 500, 1]
  const float* amps = (const float*)d_in[1];   // [B, 500, 100]
  float* out        = (float*)d_out;           // [B, 80000]
  const int B = in_sizes[0] / T_FRAMES;

  float* prefFrac = (float*)d_ws;              // B * 5000 floats (160 KB for B=8)

  harm_prefix_kernel<<<B, 64, 0, stream>>>(f0, prefFrac);
  harm_synth_kernel<<<B * BLOCKS_PER_BATCH, 128, 0, stream>>>(f0, amps, prefFrac, out);
}
